// ExpertBNN_30588757082564
// MI455X (gfx1250) — matrix operand order repaired and checked by reading, not compiled, not run on hardware
//
#include <hip/hip_runtime.h>
#include <hip/hip_bf16.h>

typedef __attribute__((ext_vector_type(16))) __bf16 v16bf;
typedef __attribute__((ext_vector_type(8)))  __bf16 v8bf;
typedef __attribute__((ext_vector_type(8)))  float  v8f;

#define EPS 1e-5f
#define LDK 168   // padded LDS row stride (bf16 elems): 336 B, 16B-aligned, conflict-free

__device__ __forceinline__ float htanh(float v) { return fminf(fmaxf(v, -1.0f), 1.0f); }
__device__ __forceinline__ float fsign(float v) { return (v > 0.0f) ? 1.0f : ((v < 0.0f) ? -1.0f : 0.0f); }

__device__ __forceinline__ v16bf cat8(v8bf lo, v8bf hi) {
    return __builtin_shufflevector(lo, hi, 0, 1, 2, 3, 4, 5, 6, 7,
                                           8, 9, 10, 11, 12, 13, 14, 15);
}

// ---------------------------------------------------------------------------
// Stage 1: fp32 conv (Cin=1, K=17, s=2, p=8) + BN + hardtanh + maxpool2
// x[256,1,16384] -> h1[256,16,4096] (bf16)
// ---------------------------------------------------------------------------
__global__ __launch_bounds__(256)
void stage1_conv_bn_pool(const float* __restrict__ x, const float* __restrict__ w1,
                         const float* __restrict__ g1, const float* __restrict__ b1,
                         const float* __restrict__ m1, const float* __restrict__ v1,
                         __bf16* __restrict__ h1) {
    const int tid = blockIdx.x * 256 + threadIdx.x;   // [0, 256*4096)
    const int b  = tid >> 12;
    const int op = tid & 4095;                        // pooled output position
    const float* xr = x + (size_t)b * 16384;

    // two conv outputs o=2*op, 2*op+1 need x[4*op-8 .. 4*op+10] (19 taps)
    float xv[19];
    const int base = 4 * op - 8;
#pragma unroll
    for (int k = 0; k < 19; ++k) {
        int p = base + k;
        xv[k] = (p >= 0 && p < 16384) ? xr[p] : 0.0f;
    }
    for (int c = 0; c < 16; ++c) {
        float s  = g1[c] * rsqrtf(v1[c] + EPS);
        float sh = b1[c] - m1[c] * s;
        float a0 = 0.0f, a1 = 0.0f;
#pragma unroll
        for (int k = 0; k < 17; ++k) {
            float w = w1[c * 17 + k];
            a0 = fmaf(xv[k],     w, a0);
            a1 = fmaf(xv[k + 2], w, a1);
        }
        a0 = htanh(fmaf(a0, s, sh));
        a1 = htanh(fmaf(a1, s, sh));
        h1[(((size_t)b * 16 + c) << 12) + op] = (__bf16)fmaxf(a0, a1);
    }
}

// ---------------------------------------------------------------------------
// Stage 2: binarized conv (Cin=16, K=9, s=2, p=4) as implicit GEMM via WMMA
// h1[256,16,4096] -> h2[256,32,1024] (bf16).  Kred=144 padded to 160.
// WG = 256 thr = 8 waves: 2 M-tiles (Cout 32) x 4 N-tiles (64 positions).
// ---------------------------------------------------------------------------
__global__ __launch_bounds__(256)
void stage2_wmma(const __bf16* __restrict__ h1, const float* __restrict__ w2,
                 const float* __restrict__ g2, const float* __restrict__ bb2,
                 const float* __restrict__ m2, const float* __restrict__ v2,
                 __bf16* __restrict__ h2) {
    __shared__ __bf16 As[32 * LDK];   // signed weights, K-padded to 160
    __shared__ __bf16 Bs[64 * LDK];   // im2col tile, K-contiguous per position
    __shared__ float  sc[32], sh[32]; // precomputed BN scale/shift
    const int b   = blockIdx.y;
    const int o0  = blockIdx.x * 64;  // conv-output base position
    const int tid = threadIdx.x;

    // per-channel BN affine, computed once per workgroup
    if (tid < 32) {
        float s = g2[tid] * rsqrtf(v2[tid] + EPS);
        sc[tid] = s;
        sh[tid] = bb2[tid] - m2[tid] * s;
    }
    // weights: sign(w2) -> +-1 bf16, zero pad K in [144,160)
    for (int idx = tid; idx < 32 * 160; idx += 256) {
        int co = idx / 160, k = idx % 160;
        float w = 0.0f;
        if (k < 144) { int ci = k / 9, kk = k % 9; w = fsign(w2[(co * 16 + ci) * 9 + kk]); }
        As[co * LDK + k] = (__bf16)w;
    }
    // im2col: Bs[n][k], k=ci*9+kk -> h1[b,ci, 2*(o0+n)-4+kk]
    const __bf16* h1b = h1 + (size_t)b * 16 * 4096;
    __builtin_prefetch(h1b + 16 * 4096, 0, 0);    // global_prefetch_b8 (next batch row)
    for (int idx = tid; idx < 64 * 160; idx += 256) {
        int n = idx / 160, k = idx % 160;
        __bf16 val = (__bf16)0.0f;
        if (k < 144) {
            int ci = k / 9, kk = k % 9;
            int pos = 2 * (o0 + n) - 4 + kk;
            if (pos >= 0 && pos < 4096) val = h1b[ci * 4096 + pos];
        }
        Bs[n * LDK + k] = val;
    }
    __syncthreads();

    const int wave = tid >> 5, lane = tid & 31;
    const int half = lane >> 4, m = lane & 15;
    const int mtile = wave & 1, ntile = wave >> 1;     // 2 x 4 tiles

    const __bf16* arow = &As[(16 * mtile + m) * LDK];  // A row M = m
    const __bf16* brow = &Bs[(16 * ntile + m) * LDK];  // B col N = m

    // preload all 5 K-chunks (20 x ds_load_b128), then chain the 5 WMMAs
    v16bf afr[5], bfr[5];
#pragma unroll
    for (int kc = 0; kc < 5; ++kc) {
        // A 16x32 bf16 layout: lane half h, elem i -> K = 8h + i + (i>=8 ? 8 : 0)
        v8bf alo = *(const v8bf*)(arow + 32 * kc + 8 * half);
        v8bf ahi = *(const v8bf*)(arow + 32 * kc + 8 * half + 16);
        afr[kc]  = cat8(alo, ahi);
        // B 32x16 layout: same K order as A
        bfr[kc]  = cat8(*(const v8bf*)(brow + 32 * kc + 8 * half),
                        *(const v8bf*)(brow + 32 * kc + 8 * half + 16));
    }
    v8f acc = {};
#pragma unroll
    for (int kc = 0; kc < 5; ++kc)
        acc = __builtin_amdgcn_wmma_f32_16x16x32_bf16(false, afr[kc], false, bfr[kc],
                                                      (short)0, acc, false, false);

    // epilogue: BN + hardtanh + maxpool2 across N (adjacent lanes)
    const int o = o0 + 16 * ntile + m;   // conv position of this lane
#pragma unroll
    for (int j = 0; j < 8; ++j) {
        int co = 16 * mtile + j + 8 * half;        // C/D layout: M = j + 8*half
        float val = htanh(fmaf(acc[j], sc[co], sh[co]));
        float pooled = fmaxf(val, __shfl_xor(val, 1, 32));
        if ((lane & 1) == 0)
            h2[((size_t)b * 32 + co) * 1024 + (o >> 1)] = (__bf16)pooled;
    }
}

// ---------------------------------------------------------------------------
// Stage 3: binarized conv (Cin=32, K=5, s=2, p=2) via WMMA, fused with
// BN + hardtanh + maxpool2 + global average pool -> feat[256,64] (f32 atomics)
// Kred = 160 exactly.  WG = 8 waves: 4 M-tiles (Cout 64) x 2 N-tiles (32 pos).
// ---------------------------------------------------------------------------
__global__ __launch_bounds__(256)
void stage3_wmma(const __bf16* __restrict__ h2, const float* __restrict__ w3,
                 const float* __restrict__ g3, const float* __restrict__ bb3,
                 const float* __restrict__ m3, const float* __restrict__ v3,
                 float* __restrict__ feat) {
    __shared__ __bf16 As[64 * LDK];
    __shared__ __bf16 Bs[32 * LDK];
    __shared__ float  sc[64], sh[64];
    const int b   = blockIdx.y;
    const int o0  = blockIdx.x * 32;
    const int tid = threadIdx.x;

    if (tid < 64) {
        float s = g3[tid] * rsqrtf(v3[tid] + EPS);
        sc[tid] = s;
        sh[tid] = bb3[tid] - m3[tid] * s;
    }
    for (int idx = tid; idx < 64 * 160; idx += 256) {
        int co = idx / 160, k = idx % 160;
        int ci = k / 5, kk = k % 5;
        As[co * LDK + k] = (__bf16)fsign(w3[(co * 32 + ci) * 5 + kk]);
    }
    const __bf16* h2b = h2 + (size_t)b * 32 * 1024;
    __builtin_prefetch(h2b + 32 * 1024, 0, 0);
    for (int idx = tid; idx < 32 * 160; idx += 256) {
        int n = idx / 160, k = idx % 160;
        int ci = k / 5, kk = k % 5;
        int pos = 2 * (o0 + n) - 2 + kk;
        __bf16 val = (__bf16)0.0f;
        if (pos >= 0 && pos < 1024) val = h2b[ci * 1024 + pos];
        Bs[n * LDK + k] = val;
    }
    __syncthreads();

    const int wave = tid >> 5, lane = tid & 31;
    const int half = lane >> 4, m = lane & 15;
    const int mtile = wave >> 1, ntile = wave & 1;    // 4 x 2 tiles

    const __bf16* arow = &As[(16 * mtile + m) * LDK];
    const __bf16* brow = &Bs[(16 * ntile + m) * LDK];

    v16bf afr[5], bfr[5];
#pragma unroll
    for (int kc = 0; kc < 5; ++kc) {
        v8bf alo = *(const v8bf*)(arow + 32 * kc + 8 * half);
        v8bf ahi = *(const v8bf*)(arow + 32 * kc + 8 * half + 16);
        afr[kc]  = cat8(alo, ahi);
        bfr[kc]  = cat8(*(const v8bf*)(brow + 32 * kc + 8 * half),
                        *(const v8bf*)(brow + 32 * kc + 8 * half + 16));
    }
    v8f acc = {};
#pragma unroll
    for (int kc = 0; kc < 5; ++kc)
        acc = __builtin_amdgcn_wmma_f32_16x16x32_bf16(false, afr[kc], false, bfr[kc],
                                                      (short)0, acc, false, false);

#pragma unroll
    for (int j = 0; j < 8; ++j) {
        int co = 16 * mtile + j + 8 * half;
        float val = htanh(fmaf(acc[j], sc[co], sh[co]));
        // maxpool pairs (value duplicated within each lane pair afterwards)
        val = fmaxf(val, __shfl_xor(val, 1, 32));
        // xor-tree sum over the 16-lane N group (each pooled value counted 2x:
        // it sits in 2 lanes after the pair max, and the tree adds each lane once)
        val += __shfl_xor(val, 2, 32);
        val += __shfl_xor(val, 4, 32);
        val += __shfl_xor(val, 8, 32);
        val += __shfl_xor(val, 1, 32);
        if (m == 0)   // /2 duplication, /512 (half of the /256 mean)
            atomicAdd(&feat[b * 64 + co], val * (1.0f / 1024.0f));
    }
}

// ---------------------------------------------------------------------------
// Head: concat(feat, rms) -> fc1(65->32)+hardtanh -> fc2(32->2)
// ---------------------------------------------------------------------------
__global__ __launch_bounds__(256)
void head_fc(const float* __restrict__ feat, const float* __restrict__ rms,
             const float* __restrict__ fc1w, const float* __restrict__ fc1b,
             const float* __restrict__ fc2w, const float* __restrict__ fc2b,
             float* __restrict__ out) {
    int b = blockIdx.x * blockDim.x + threadIdx.x;
    if (b >= 256) return;
    float c[65];
#pragma unroll
    for (int j = 0; j < 64; ++j) c[j] = feat[b * 64 + j];
    c[64] = rms[b];
    float h[32];
    for (int i = 0; i < 32; ++i) {
        float a = fc1b[i];
#pragma unroll
        for (int j = 0; j < 65; ++j) a = fmaf(c[j], fc1w[i * 65 + j], a);
        h[i] = htanh(a);
    }
    for (int o = 0; o < 2; ++o) {
        float a = fc2b[o];
#pragma unroll
        for (int i = 0; i < 32; ++i) a = fmaf(h[i], fc2w[o * 32 + i], a);
        out[b * 2 + o] = a;
    }
}

// ---------------------------------------------------------------------------
extern "C" void kernel_launch(void* const* d_in, const int* in_sizes, int n_in,
                              void* d_out, int out_size, void* d_ws, size_t ws_size,
                              hipStream_t stream) {
    const float* x    = (const float*)d_in[0];
    const float* rms  = (const float*)d_in[1];
    const float* w1   = (const float*)d_in[2];
    const float* g1   = (const float*)d_in[3];
    const float* b1   = (const float*)d_in[4];
    const float* m1   = (const float*)d_in[5];
    const float* v1   = (const float*)d_in[6];
    const float* w2   = (const float*)d_in[7];
    const float* g2   = (const float*)d_in[8];
    const float* b2   = (const float*)d_in[9];
    const float* m2   = (const float*)d_in[10];
    const float* v2   = (const float*)d_in[11];
    const float* w3   = (const float*)d_in[12];
    const float* g3   = (const float*)d_in[13];
    const float* b3   = (const float*)d_in[14];
    const float* m3   = (const float*)d_in[15];
    const float* v3   = (const float*)d_in[16];
    const float* fc1w = (const float*)d_in[17];
    const float* fc1b = (const float*)d_in[18];
    const float* fc2w = (const float*)d_in[19];
    const float* fc2b = (const float*)d_in[20];

    char* ws = (char*)d_ws;
    const size_t h1_bytes = (size_t)256 * 16 * 4096 * sizeof(__bf16);   // 32 MB
    const size_t h2_bytes = (size_t)256 * 32 * 1024 * sizeof(__bf16);   // 16 MB
    __bf16* h1   = (__bf16*)ws;
    __bf16* h2   = (__bf16*)(ws + h1_bytes);
    float*  feat = (float*)(ws + h1_bytes + h2_bytes);                  // 64 KB

    stage1_conv_bn_pool<<<4096, 256, 0, stream>>>(x, w1, g1, b1, m1, v1, h1);
    stage2_wmma<<<dim3(32, 256), 256, 0, stream>>>(h1, w2, g2, b2, m2, v2, h2);
    hipMemsetAsync(feat, 0, (size_t)256 * 64 * sizeof(float), stream);
    stage3_wmma<<<dim3(16, 256), 256, 0, stream>>>(h2, w3, g3, b3, m3, v3, feat);
    head_fc<<<1, 256, 0, stream>>>(feat, rms, fc1w, fc1b, fc2w, fc2b, (float*)d_out);
}
